// GrantConv_78271484002572
// MI455X (gfx1250) — matrix operand order repaired and checked by reading, not compiled, not run on hardware
//
#include <hip/hip_runtime.h>

#define NN 8192      // nodes
#define NF 256       // features
#define NE 131072    // edges
#define ROWH 40      // padded LDS row stride in halves (80 B) -> conflict-free banks
#define SLICE_B (ROWH * NF * 2)   // 20480 B per K-slice buffer

typedef __attribute__((ext_vector_type(16))) _Float16 v16h;
typedef __attribute__((ext_vector_type(8)))  _Float16 v8h;
typedef __attribute__((ext_vector_type(8)))  float    v8f;
typedef __attribute__((ext_vector_type(4)))  unsigned int u32x4;

// ---------------- zero the f16 adjacency (128 MB) ----------------
__global__ __launch_bounds__(256) void zero_kernel(u32x4* __restrict__ p, long long n) {
  long long i = (long long)blockIdx.x * blockDim.x + threadIdx.x;
  if (i < n) { u32x4 z = {0u, 0u, 0u, 0u}; p[i] = z; }
}

// ---------------- adj[src,dst] = w (scatter set, plain stores) ----------------
__global__ __launch_bounds__(256) void adj_set_kernel(const long long* __restrict__ ei,
                                                      const float* __restrict__ w,
                                                      _Float16* __restrict__ A) {
  int e = blockIdx.x * blockDim.x + threadIdx.x;
  if (e >= NE) return;
  int s = (int)ei[e];
  int d = (int)ei[NE + e];
  A[(size_t)s * NN + d] = (_Float16)w[e];
}

// ---------------- adj[dst,src] += w via global_atomic_pk_add_f16 ----------------
__global__ __launch_bounds__(256) void adj_add_kernel(const long long* __restrict__ ei,
                                                      const float* __restrict__ w,
                                                      _Float16* __restrict__ A) {
  int e = blockIdx.x * blockDim.x + threadIdx.x;
  if (e >= NE) return;
  int s = (int)ei[e];
  int d = (int)ei[NE + e];
  size_t idx = (size_t)d * NN + s;
  union { _Float16 h; unsigned short u; } cv;
  cv.h = (_Float16)w[e];
  unsigned int packed = (idx & 1) ? ((unsigned int)cv.u << 16) : (unsigned int)cv.u;
  unsigned long long addr = (unsigned long long)(A + (idx & ~(size_t)1));
  asm volatile("global_atomic_pk_add_f16 %0, %1, off scope:SCOPE_DEV"
               :: "v"(addr), "v"(packed) : "memory");
}

// ---- copy x into out[:,0:256] and build transposed f16 H^T (256 x 8192) ----
__global__ __launch_bounds__(256) void prep_kernel(const float* __restrict__ x,
                                                   float* __restrict__ out,
                                                   _Float16* __restrict__ Ht) {
  int id = blockIdx.x * blockDim.x + threadIdx.x;   // 0 .. NN*NF-1
  int m = id >> 8;            // node
  int n = id & (NF - 1);      // feature
  float v = x[id];
  out[(size_t)m * (4 * NF) + n] = v;
  Ht[(size_t)n * NN + m] = (_Float16)v;
}

// Stage one 32 x 256 B^T K-slice (16 KB of data, padded rows in LDS) via async
// copies. Each of 128 threads moves rows n=2*tid, 2*tid+1: a row is 64 B = 4
// async b128 ops (INST_OFFSET applies to BOTH the global and LDS addresses, so
// the padded LDS layout just uses a per-row base of n*80 B).
__device__ __forceinline__ void stage_slice(const _Float16* __restrict__ Bt,
                                            int k0, unsigned lds_byte_base, int tid) {
#pragma unroll
  for (int q = 0; q < 2; ++q) {
    int n = tid * 2 + q;
    unsigned long long ga = (unsigned long long)(Bt + (size_t)n * NN + k0);
    unsigned la = lds_byte_base + (unsigned)n * (ROWH * 2);
    asm volatile("global_load_async_to_lds_b128 %0, %1, off"
                 :: "v"(la), "v"(ga) : "memory");
    asm volatile("global_load_async_to_lds_b128 %0, %1, off offset:16"
                 :: "v"(la), "v"(ga) : "memory");
    asm volatile("global_load_async_to_lds_b128 %0, %1, off offset:32"
                 :: "v"(la), "v"(ga) : "memory");
    asm volatile("global_load_async_to_lds_b128 %0, %1, off offset:48"
                 :: "v"(la), "v"(ga) : "memory");
  }
}

// ---------------- out = A(8192x8192 f16) @ H  via WMMA f16->f32 ----------------
// Bt = H^T (NF x NN, f16). Wave: 16 rows x 256 cols (16 accumulators).
// WG = 4 waves = 64 rows. B K-slices triple-buffered in LDS via async-to-LDS.
__global__ __launch_bounds__(128)
void gemm_kernel(const _Float16* __restrict__ A,
                 const _Float16* __restrict__ Bt,
                 float* __restrict__ out,        // pre-offset to hop's column base
                 _Float16* __restrict__ BtOut) { // transposed f16 result for next hop
  __shared__ _Float16 Bs[3 * ROWH * NF];        // 3 x 20 KB padded slices
  const int tid  = threadIdx.x;
  const int lane = tid & 31;
  const int wave = tid >> 5;
  const int lm   = lane & 15;
  const int g    = lane >> 4;                  // lane group (K-half select)
  const int m0   = (blockIdx.x * 4 + wave) * 16;

  const _Float16* arow = A + (size_t)(m0 + lm) * NN;
  const unsigned lds0 = (unsigned)(size_t)&Bs[0];   // low 32 bits = LDS byte offset

  v8f acc[16];
#pragma unroll
  for (int t = 0; t < 16; ++t) { v8f z = {0, 0, 0, 0, 0, 0, 0, 0}; acc[t] = z; }

  // Prologue: slices 0 and 1 in flight (8 async ops per slice per wave).
  stage_slice(Bt, 0, lds0, tid);
  stage_slice(Bt, 32, lds0 + (unsigned)SLICE_B, tid);

  for (int i = 0; i < NN / 32; ++i) {
    const int k0 = i * 32;
    // Ensure slice i has landed (slice i+1 may remain in flight: 8 ops/wave).
    if (i < NN / 32 - 1) asm volatile("s_wait_asynccnt 0x8" ::: "memory");
    else                 asm volatile("s_wait_asynccnt 0x0" ::: "memory");
    __syncthreads();
    // Prefetch slice i+2 into buffer (i+2)%3 == (i-1)%3; its previous contents
    // (slice i-1) were fully consumed before every wave signaled this barrier.
    if (i + 2 < NN / 32)
      stage_slice(Bt, k0 + 64, lds0 + (unsigned)((i + 2) % 3) * (unsigned)SLICE_B, tid);

    const _Float16* Bp = Bs + (i % 3) * (ROWH * NF);

    // A fragment: lane holds row m0+lm, K chunk [16g,16g+16)
    v8h a_lo = *(const v8h*)(arow + k0 + 16 * g);
    v8h a_hi = *(const v8h*)(arow + k0 + 16 * g + 8);
    v16h a = __builtin_shufflevector(a_lo, a_hi,
                                     0, 1, 2, 3, 4, 5, 6, 7,
                                     8, 9, 10, 11, 12, 13, 14, 15);

    // Software-pipelined tile loop: load fragment t+1 before issuing wmma t.
    v16h b = *(const v16h*)(Bp + lm * ROWH + 16 * g);
#pragma unroll
    for (int t = 0; t < 16; ++t) {
      v16h bn = b;
      if (t < 15) bn = *(const v16h*)(Bp + ((t + 1) * 16 + lm) * ROWH + 16 * g);
      acc[t] = __builtin_amdgcn_wmma_f32_16x16x32_f16(
          false, a, false, b, (short)0, acc[t], false, false);
      b = bn;
    }
  }

  // C/D layout: element v -> row m0 + 8g + v, col t*16 + lm
#pragma unroll
  for (int t = 0; t < 16; ++t) {
    int n = t * 16 + lm;
    int mb = m0 + 8 * g;
    v8h hv;
#pragma unroll
    for (int v = 0; v < 8; ++v) {
      float f = acc[t][v];
      out[(size_t)(mb + v) * (4 * NF) + n] = f;
      hv[v] = (_Float16)f;
    }
    *(v8h*)(BtOut + (size_t)n * NN + mb) = hv;  // contiguous 8 rows -> b128 store
  }
}

extern "C" void kernel_launch(void* const* d_in, const int* in_sizes, int n_in,
                              void* d_out, int out_size, void* d_ws, size_t ws_size,
                              hipStream_t stream) {
  (void)in_sizes; (void)n_in; (void)out_size;
  // d_in order: k (scalar), x (f32), edge_index (i64), edge_weight (f32)
  const float* x = (const float*)d_in[1];
  const long long* ei = (const long long*)d_in[2];
  const float* w = (const float*)d_in[3];
  float* out = (float*)d_out;

  const size_t a_bytes = (size_t)NN * NN * sizeof(_Float16);  // 128 MB
  const size_t h_elems = (size_t)NF * NN;                     // 2M f16 = 4 MB
  if (ws_size < a_bytes + 2 * h_elems * sizeof(_Float16)) return;

  _Float16* Af  = (_Float16*)d_ws;
  _Float16* HtA = (_Float16*)((char*)d_ws + a_bytes);
  _Float16* HtB = HtA + h_elems;

  long long n16 = (long long)(a_bytes / 16);
  zero_kernel<<<(unsigned)((n16 + 255) / 256), 256, 0, stream>>>((u32x4*)Af, n16);
  adj_set_kernel<<<NE / 256, 256, 0, stream>>>(ei, w, Af);
  adj_add_kernel<<<NE / 256, 256, 0, stream>>>(ei, w, Af);
  prep_kernel<<<(NN * NF) / 256, 256, 0, stream>>>(x, out, HtA);

  // hop j: out[:, j*256:(j+1)*256] = A @ H_{j-1}, and H_j^T (f16) for next hop
  gemm_kernel<<<NN / 64, 128, 0, stream>>>(Af, HtA, out + 1 * NF, HtB);
  gemm_kernel<<<NN / 64, 128, 0, stream>>>(Af, HtB, out + 2 * NF, HtA);
  gemm_kernel<<<NN / 64, 128, 0, stream>>>(Af, HtA, out + 3 * NF, HtB);
}
